// interattention_29240137351727
// MI455X (gfx1250) — matrix operand order repaired and checked by reading, not compiled, not run on hardware
//
#include <hip/hip_runtime.h>

// ---------------------------------------------------------------------------
// Fused dual-softmax cross-attention for MI455X (gfx1250, wave32, WMMA).
//   out[b,p] = conv * sum_c gate[c] * ( sigmoid(x1_att)[c,p]*x1[c,p]
//                                     + sigmoid(x2_att)[c,p]*x2[c,p] )
// Flash-style: A = X1^T X2 is never materialized. For (P,Q):
//   att[c,m] = (sum_n P[c,n] * e^{A[n,m]}) / (sum_n e^{A[n,m]}),  A = P^T Q
// launch #1 (P=x1,Q=x2) -> x2_att/out2 part; launch #2 swapped -> x1_att/out1.
// All GEMMs via v_wmma_f32_16x16x32_bf16; operand fetches are batched per
// K-step so the compiler can merge s_wait_dscnt and issue WMMAs back-to-back.
// ---------------------------------------------------------------------------

#define BATCH   4
#define CD      128     // channels (K of GEMM1)
#define HW      4096    // H*W
#define TM      64      // m-columns owned per block
#define TN      64      // n-chunk per loop iteration (K of GEMM2)
#define THREADS 256
#define NW      8       // waves per block (wave32)

typedef __attribute__((ext_vector_type(16))) __bf16 bf16x16;
typedef __attribute__((ext_vector_type(8)))  float  f32x8;

struct U32x8 { uint4 lo, hi; };
static_assert(sizeof(U32x8) == sizeof(bf16x16), "frag size mismatch");

__device__ __forceinline__ unsigned short f2bf(float f) {
  unsigned u = __float_as_uint(f);
  u += 0x7FFFu + ((u >> 16) & 1u);        // round-to-nearest-even
  return (unsigned short)(u >> 16);
}

// XOR swizzle of 16B granules inside a K-innermost LDS row: conflict-free
// ds_load_b128 operand fetches without padding.
__device__ __forceinline__ int swz_idx(int row, int k, int mask) {
  return (((k >> 3) ^ (row & mask)) << 3) | (k & 7);
}

// A operand (16x32 bf16): lane L -> row L%16; holds
// K = kc0 + 16*(L/16) + 0..15 (matches B).  -> two b128 LDS loads.
template<int K, int MASK>
__device__ __forceinline__ bf16x16 load_afrag(const unsigned short* base,
                                              int row0, int kc0, int lane) {
  const int row = row0 + (lane & 15);
  const int h   = lane >> 4;
  const unsigned short* rp = base + row * K;
  const int c0 = (kc0 >> 3) + 2 * h;
  U32x8 u;
  u.lo = *(const uint4*)(rp + ((( c0     ) ^ (row & MASK)) << 3));
  u.hi = *(const uint4*)(rp + (((c0 + 1) ^ (row & MASK)) << 3));
  return __builtin_bit_cast(bf16x16, u);
}

// B operand (32x16 bf16): lane L -> col L%16; holds K = kc0 + 16*(L/16)+0..15.
template<int K, int MASK>
__device__ __forceinline__ bf16x16 load_bfrag(const unsigned short* base,
                                              int col0, int kc0, int lane) {
  const int col = col0 + (lane & 15);
  const int h   = lane >> 4;
  const unsigned short* rp = base + col * K;
  const int c0 = (kc0 >> 3) + 2 * h;
  U32x8 u;
  u.lo = *(const uint4*)(rp + ((( c0     ) ^ (col & MASK)) << 3));
  u.hi = *(const uint4*)(rp + (((c0 + 1) ^ (col & MASK)) << 3));
  return __builtin_bit_cast(bf16x16, u);
}

__device__ __forceinline__ f32x8 wmma_bf16(bf16x16 a, bf16x16 b, f32x8 c) {
  return __builtin_amdgcn_wmma_f32_16x16x32_bf16(
      /*neg_a=*/false, a, /*neg_b=*/false, b,
      /*c_mod=*/(short)0, c, /*reuse_a=*/false, /*reuse_b=*/false);
}

__global__ __launch_bounds__(THREADS)
void fused_cross_attn(const float* __restrict__ P, const float* __restrict__ Q,
                      const float* __restrict__ gate, const float* __restrict__ conv,
                      float* __restrict__ out) {
  __shared__ __align__(16) unsigned short sPnk[TN * CD]; // P tile, row=n, K=c (GEMM1 A)
  __shared__ __align__(16) unsigned short sPkn[CD * TN]; // P tile, row=c, K=n (GEMM2 A)
  __shared__ __align__(16) unsigned short sQ  [TM * CD]; // Q tile, row=m, K=c (GEMM1 B)
  __shared__ __align__(16) unsigned short sE  [TM * TN]; // exp(A) , row=m, K=n (GEMM2 B)
  __shared__ float sCS [4 * TM];   // per-n-strip column-sum partials (fixed order)
  __shared__ float sOut[NW * TM];  // per-wave output partials
  __shared__ float sGate[CD];

  const int b    = blockIdx.y;
  const int m0   = blockIdx.x * TM;
  const int t    = threadIdx.x;
  const int lane = t & 31;
  const int w    = t >> 5;
  const int hh   = lane >> 4;
  const int l16  = lane & 15;
  const int s    = w >> 1;          // GEMM1: n-strip (rows 16s..16s+15)
  const int q    = w & 1;           // GEMM1: m half (cols 32q..32q+31)

  const float* Pb = P + (size_t)b * CD * HW;
  const float* Qb = Q + (size_t)b * CD * HW;

  // Stage Q tile (persistent), gate weights, zero partial sums.
  for (int idx = t; idx < TM * CD; idx += THREADS) {
    int m = idx % TM, c = idx / TM;
    sQ[m * CD + swz_idx(m, c, 15)] = f2bf(Qb[c * HW + m0 + m]);
  }
  if (t < CD) sGate[t] = gate[t];
  sCS[t] = 0.0f;                    // 4*TM == THREADS

  f32x8 acc2[4] = {};               // x_att accumulator: c-strip 16w x (4 m-tiles)
  __syncthreads();

  for (int n0 = 0; n0 < HW; n0 += TN) {
    // ---- stage P n-chunk in both K-layouts (fp32 -> bf16) ----
    for (int idx = t; idx < TN * CD; idx += THREADS) {
      int n = idx % TN, c = idx / TN;
      unsigned short v = f2bf(Pb[c * HW + n0 + n]);
      sPnk[n * CD + swz_idx(n, c, 15)] = v;
      sPkn[c * TN + swz_idx(c, n, 7)]  = v;
    }
    if (n0 + TN < HW) {             // pull next chunk toward L2/L0
      __builtin_prefetch(Pb + (t & 127) * HW + (n0 + TN) + ((t >> 7) * 32), 0, 1);
    }
    __syncthreads();

    // ---- GEMM1: A_chunk[TN x TM] = P^T * Q over K=C ----
    // Batch all operand fetches of a K-step, then issue both WMMAs.
    f32x8 acc1[2] = {};
    #pragma unroll
    for (int kc = 0; kc < CD; kc += 32) {
      bf16x16 af = load_afrag<CD, 15>(sPnk, 16 * s, kc, lane);
      bf16x16 b0 = load_bfrag<CD, 15>(sQ, 32 * q,      kc, lane);
      bf16x16 b1 = load_bfrag<CD, 15>(sQ, 32 * q + 16, kc, lane);
      acc1[0] = wmma_bf16(af, b0, acc1[0]);
      acc1[1] = wmma_bf16(af, b1, acc1[1]);
    }
    // ---- exp, stage E as GEMM2 B operand, accumulate column sums ----
    #pragma unroll
    for (int j = 0; j < 2; ++j) {
      const int ml = 32 * q + 16 * j + l16;
      float csum = 0.0f;
      #pragma unroll
      for (int r = 0; r < 8; ++r) {   // C/D layout: (n = 16s + 8*hh + r, m = ml)
        float e = __expf(acc1[j][r]);
        csum += e;
        int nl = 16 * s + 8 * hh + r;
        sE[ml * TN + swz_idx(ml, nl, 7)] = f2bf(e);
      }
      csum += __shfl_xor(csum, 16, 32);              // combine lane halves
      if (lane < 16) sCS[s * TM + ml] += csum;       // unique (s,ml) owner: no race
    }
    __syncthreads();

    // ---- GEMM2: acc2 += P[c, n-chunk] * E[n-chunk, m] over K=TN ----
    // Hoist A + all four B fragments: one wait, four back-to-back WMMAs.
    #pragma unroll
    for (int kc = 0; kc < TN; kc += 32) {
      bf16x16 af = load_afrag<TN, 7>(sPkn, 16 * w, kc, lane);
      bf16x16 b0 = load_bfrag<TN, 7>(sE,  0, kc, lane);
      bf16x16 b1 = load_bfrag<TN, 7>(sE, 16, kc, lane);
      bf16x16 b2 = load_bfrag<TN, 7>(sE, 32, kc, lane);
      bf16x16 b3 = load_bfrag<TN, 7>(sE, 48, kc, lane);
      acc2[0] = wmma_bf16(af, b0, acc2[0]);
      acc2[1] = wmma_bf16(af, b1, acc2[1]);
      acc2[2] = wmma_bf16(af, b2, acc2[2]);
      acc2[3] = wmma_bf16(af, b3, acc2[3]);
    }
    __syncthreads();   // sE/sPnk/sPkn free for next iteration
  }

  // ---- normalize, sigmoid, gate-contract, emit output tile ----
  const float cw = conv[0];
  #pragma unroll
  for (int mt = 0; mt < 4; ++mt) {
    const int ml = 16 * mt + l16;
    const float cs = sCS[0 * TM + ml] + sCS[1 * TM + ml]
                   + sCS[2 * TM + ml] + sCS[3 * TM + ml];
    const float rcs = 1.0f / cs;
    float osum = 0.0f;
    #pragma unroll
    for (int r = 0; r < 8; ++r) {       // C/D layout: c = 16w + 8*hh + r
      const int   cl   = 16 * w + 8 * hh + r;
      const float att  = acc2[mt][r] * rcs;
      const float mask = 1.0f / (1.0f + __expf(-att));
      osum += sGate[cl] * mask * Qb[cl * HW + m0 + ml];
    }
    osum += __shfl_xor(osum, 16, 32);
    if (lane < 16) sOut[w * TM + ml] = osum;
  }
  __syncthreads();
  if (t < TM) {
    float v = 0.0f;
    #pragma unroll
    for (int i = 0; i < NW; ++i) v += sOut[i * TM + t];
    // Two stream-ordered launches add their halves; d_out was memset to 0.
    out[(size_t)b * HW + m0 + t] += cw * v;
  }
}

extern "C" void kernel_launch(void* const* d_in, const int* in_sizes, int n_in,
                              void* d_out, int out_size, void* d_ws, size_t ws_size,
                              hipStream_t stream) {
  (void)in_sizes; (void)n_in; (void)d_ws; (void)ws_size;
  const float* x1   = (const float*)d_in[0];
  const float* x2   = (const float*)d_in[1];
  const float* gate = (const float*)d_in[2];
  const float* conv = (const float*)d_in[3];
  float* out = (float*)d_out;

  hipMemsetAsync(out, 0, (size_t)out_size * sizeof(float), stream);

  dim3 grid(HW / TM, BATCH);   // 64 m-tiles x 4 batches = 256 blocks
  // out2 part: x2_att from A = x1^T x2 (column softmax)
  fused_cross_attn<<<grid, THREADS, 0, stream>>>(x1, x2, gate, conv, out);
  // out1 part: x1_att from A^T (row softmax), same kernel with operands swapped
  fused_cross_attn<<<grid, THREADS, 0, stream>>>(x2, x1, gate, conv, out);
}
